// MultiHeadLatentAttention_13443247636571
// MI455X (gfx1250) — matrix operand order repaired and checked by reading, compiled, not run on hardware
//
#include <hip/hip_runtime.h>

// ---------------------------------------------------------------------------
// MLA forward for MI455X (gfx1250, wave32, WMMA 16x16x32 f16)
//   cvt(X, all W) -> f16
//   GEMM1: qd   = Xh @ WqdT + bqd            (4096x1536, K=2048)
//   rms_q: qn_h = rms(qd)*gq  (f16)
//   GEMM2: q    = qn_h @ WquT + bqu          (4096x3072, K=1536)
//   rope_q: qf_h = [qNope | rope(qPos)] * scale  (f16, [m][h][192])
//   GEMM3: kv   = Xh @ WkvT + bkv            (4096x576,  K=2048)
//   kv_post: kvn_h = rms(kv[:512])*gkv ; kRot -> kf_h[...,128:192]
//   GEMM4: kvu  = kvn_h @ WkvuT + bkvu       (4096x4096, K=512)
//   build_kv: kf_h[...,0:128]=kNope ; v_h = v
//   attn: flash attention (online softmax) -> ao_h (f16)
//   GEMM5: out  = ao_h @ WoT + bo -> d_out (f32)
// Async global->LDS (ASYNCcnt) staging where the toolchain provides it.
// ---------------------------------------------------------------------------

typedef _Float16 v16h __attribute__((ext_vector_type(16)));
typedef _Float16 v8h  __attribute__((ext_vector_type(8)));
typedef float    v8f  __attribute__((ext_vector_type(8)));

#define WMMA_F32_F16(a, b, c) \
  __builtin_amdgcn_wmma_f32_16x16x32_f16(false, (a), false, (b), (short)0, (c), false, false)

#if defined(__has_builtin)
#  if __has_builtin(__builtin_amdgcn_global_load_async_to_lds_b128)
#    define USE_ASYNC_COPY 1
#  endif
#endif
#ifndef USE_ASYNC_COPY
#  define USE_ASYNC_COPY 0
#endif

#if USE_ASYNC_COPY
// Builtin signature (from toolchain diagnostics):
//   (v4i addrspace(1)* gsrc, v4i addrspace(3)* ldst, imm offset, imm cpol)
typedef int vi4_t __attribute__((vector_size(16)));
typedef __attribute__((address_space(1))) vi4_t* gvi4_p;
typedef __attribute__((address_space(3))) vi4_t* lvi4_p;
// direct memory->LDS DMA, tracked by ASYNCcnt (no VGPR round trip)
#define ASYNC_G2L_B128(gsrc, ldst)                                      \
  __builtin_amdgcn_global_load_async_to_lds_b128(                       \
      (gvi4_p)(size_t)(gsrc), (lvi4_p)(ldst), 0, 0)
#define WAIT_ASYNC() __asm__ volatile("s_wait_asynccnt 0x0" ::: "memory")
#else
#define WAIT_ASYNC() do {} while (0)
#endif

// A-fragment (16x32 f16): lane holds row lane%16; halves at +0/+16 halfs.
__device__ __forceinline__ v16h ld_frag_split(const _Float16* p) {
  v8h lo = *(const v8h*)p;
  v8h hi = *(const v8h*)(p + 16);
  return __builtin_shufflevector(lo, hi, 0,1,2,3,4,5,6,7,8,9,10,11,12,13,14,15);
}
// B-fragment (32x16 f16): lane holds col lane%16; 16 contiguous K halfs.
__device__ __forceinline__ v16h ld_frag_contig(const _Float16* p) {
  v8h lo = *(const v8h*)p;
  v8h hi = *(const v8h*)(p + 8);
  return __builtin_shufflevector(lo, hi, 0,1,2,3,4,5,6,7,8,9,10,11,12,13,14,15);
}
__device__ __forceinline__ v8f zero_v8f() {
  v8f z;
#pragma unroll
  for (int i = 0; i < 8; ++i) z[i] = 0.f;
  return z;
}

// ---------------------------------------------------------------------------
__global__ void cvt_f32_f16(const float* __restrict__ src, _Float16* __restrict__ dst, int n) {
  int i = blockIdx.x * 256 + threadIdx.x;
  if (i < n) dst[i] = (_Float16)src[i];
}

// ---------------------------------------------------------------------------
// WMMA GEMM:  C[M,N] = A[M,K](f16,row-major) * W[N,K]^T (f16,row-major) + bias
// Block tile 128x128, BK=32, 8 waves (4x2), wave tile 32x64, double-buffered LDS.
// ---------------------------------------------------------------------------
__global__ __launch_bounds__(256)
void gemm_wmma(const _Float16* __restrict__ A, const _Float16* __restrict__ Bw,
               const float* __restrict__ bias, float* __restrict__ C,
               int M, int N, int K) {
  const int t = threadIdx.x, lane = t & 31, wid = t >> 5;
  const int half = lane >> 4, lm = lane & 15;
  const int m0 = blockIdx.y * 128, n0 = blockIdx.x * 128;
  const int wm = (wid >> 1) * 32, wn = (wid & 1) * 64;

  __shared__ __align__(16) _Float16 As[2][128 * 40];
  __shared__ __align__(16) _Float16 Bs[2][128 * 40];

  const int lrow = t >> 1;        // tile row this thread stages (2 threads/row)
  const int lcol = (t & 1) * 16;  // 16 halfs each
  const size_t a_off = (size_t)(m0 + lrow) * K + lcol;
  const int brow_g = n0 + lrow;
  const bool bvalid = brow_g < N;
  const size_t b_off = (size_t)(bvalid ? brow_g : 0) * K + lcol;

  // OOB weight rows: write zeros once; never touched again by async fills.
  if (!bvalid) {
    const uint4 z = make_uint4(0, 0, 0, 0);
#pragma unroll
    for (int sb = 0; sb < 2; ++sb) {
      *(uint4*)(&Bs[sb][lrow * 40 + lcol])     = z;
      *(uint4*)(&Bs[sb][lrow * 40 + lcol + 8]) = z;
    }
  }

  auto load_tile = [&](int kk, int sb) {
#if USE_ASYNC_COPY
    ASYNC_G2L_B128(A + a_off + kk,     &As[sb][lrow * 40 + lcol]);
    ASYNC_G2L_B128(A + a_off + kk + 8, &As[sb][lrow * 40 + lcol + 8]);
    if (bvalid) {
      ASYNC_G2L_B128(Bw + b_off + kk,     &Bs[sb][lrow * 40 + lcol]);
      ASYNC_G2L_B128(Bw + b_off + kk + 8, &Bs[sb][lrow * 40 + lcol + 8]);
    }
#else
    uint4 a0 = *(const uint4*)(A + a_off + kk);
    uint4 a1 = *(const uint4*)(A + a_off + kk + 8);
    *(uint4*)(&As[sb][lrow * 40 + lcol])     = a0;
    *(uint4*)(&As[sb][lrow * 40 + lcol + 8]) = a1;
    if (bvalid) {
      uint4 b0 = *(const uint4*)(Bw + b_off + kk);
      uint4 b1 = *(const uint4*)(Bw + b_off + kk + 8);
      *(uint4*)(&Bs[sb][lrow * 40 + lcol])     = b0;
      *(uint4*)(&Bs[sb][lrow * 40 + lcol + 8]) = b1;
    }
#endif
  };

  v8f acc[2][4];
#pragma unroll
  for (int im = 0; im < 2; ++im)
#pragma unroll
    for (int in = 0; in < 4; ++in) acc[im][in] = zero_v8f();

  load_tile(0, 0);
  const int KT = K >> 5;
  int s = 0;
  for (int kt = 0; kt < KT; ++kt) {
    WAIT_ASYNC();            // our async fills from last iteration have landed
    __syncthreads();         // everyone's fills visible
    if (kt + 2 < KT) {
      __builtin_prefetch(A + a_off + (size_t)(kt + 2) * 32, 0, 1);   // global_prefetch_b8
      __builtin_prefetch(Bw + b_off + (size_t)(kt + 2) * 32, 0, 1);
    }
    if (kt + 1 < KT) load_tile((kt + 1) * 32, s ^ 1);   // overlaps with compute below

    v16h af[2], bf[4];
#pragma unroll
    for (int im = 0; im < 2; ++im)
      af[im] = ld_frag_contig(&As[s][(wm + im * 16 + lm) * 40 + half * 16]);
#pragma unroll
    for (int in = 0; in < 4; ++in)
      bf[in] = ld_frag_contig(&Bs[s][(wn + in * 16 + lm) * 40 + half * 16]);
#pragma unroll
    for (int im = 0; im < 2; ++im)
#pragma unroll
      for (int in = 0; in < 4; ++in)
        acc[im][in] = WMMA_F32_F16(af[im], bf[in], acc[im][in]);
    s ^= 1;
  }

#pragma unroll
  for (int im = 0; im < 2; ++im) {
    const int row_g = m0 + wm + im * 16 + half * 8;  // rows row_g..row_g+7
#pragma unroll
    for (int in = 0; in < 4; ++in) {
      const int col = n0 + wn + in * 16 + lm;
      if (col < N) {
        const float bsv = bias[col];
#pragma unroll
        for (int i = 0; i < 8; ++i)
          C[(size_t)(row_g + i) * N + col] = acc[im][in][i] + bsv;
      }
    }
  }
}

// ---------------------------------------------------------------------------
__global__ void rms_rows(const float* __restrict__ x, const float* __restrict__ g,
                         _Float16* __restrict__ out, int ncols) {
  const int row = blockIdx.x, t = threadIdx.x;
  const float* xr = x + (size_t)row * ncols;
  float ss = 0.f;
  for (int c = t; c < ncols; c += 256) { float v = xr[c]; ss += v * v; }
  __shared__ float red[256];
  red[t] = ss; __syncthreads();
  for (int s = 128; s > 0; s >>= 1) { if (t < s) red[t] += red[t + s]; __syncthreads(); }
  const float inv = rsqrtf(red[0] / (float)ncols + 1e-6f);
  _Float16* orow = out + (size_t)row * ncols;
  for (int c = t; c < ncols; c += 256) orow[c] = (_Float16)(xr[c] * inv * g[c]);
}

// ---------------------------------------------------------------------------
__global__ void rope_q(const float* __restrict__ q, const float* __restrict__ bf,
                       _Float16* __restrict__ qf) {
  const int idx = blockIdx.x * 256 + threadIdx.x;      // < 4096*16*192
  const int j = idx % 192, mh = idx / 192;
  const int m = mh >> 4, s = m & 1023;
  const float scale = 0.07216878364870323f;            // 192^-0.5
  const float* qr = q + (size_t)mh * 192;
  float val;
  if (j < 128) {
    val = qr[j];
  } else {
    const int jr = j - 128;
    const float ang = bf[s * 64 + jr];
    const float qp = qr[128 + jr];
    const float other = (jr < 32) ? -qr[128 + jr + 32] : qr[128 + jr - 32];
    val = qp * __cosf(ang) + other * __sinf(ang);
  }
  qf[idx] = (_Float16)(val * scale);
}

// ---------------------------------------------------------------------------
__global__ void kv_post(const float* __restrict__ kv, const float* __restrict__ gkv,
                        const float* __restrict__ bf,
                        _Float16* __restrict__ kvn, _Float16* __restrict__ kf) {
  const int m = blockIdx.x, t = threadIdx.x, s = m & 1023;
  const float* kr = kv + (size_t)m * 576;
  float ss = 0.f;
  for (int c = t; c < 512; c += 256) { float v = kr[c]; ss += v * v; }
  __shared__ float red[256];
  red[t] = ss; __syncthreads();
  for (int sr = 128; sr > 0; sr >>= 1) { if (t < sr) red[t] += red[t + sr]; __syncthreads(); }
  const float inv = rsqrtf(red[0] / 512.f + 1e-6f);
  for (int c = t; c < 512; c += 256)
    kvn[(size_t)m * 512 + c] = (_Float16)(kr[c] * inv * gkv[c]);
  if (t < 64) {
    const int j = t;
    const float ang = bf[s * 64 + j];
    const float kp = kr[512 + j];
    const float other = (j < 32) ? -kr[512 + j + 32] : kr[512 + j - 32];
    const _Float16 hv = (_Float16)(kp * __cosf(ang) + other * __sinf(ang));
#pragma unroll
    for (int h = 0; h < 16; ++h)
      kf[((size_t)m * 16 + h) * 192 + 128 + j] = hv;
  }
}

// ---------------------------------------------------------------------------
__global__ void build_kv(const float* __restrict__ kvu,
                         _Float16* __restrict__ kf, _Float16* __restrict__ vh) {
  const int idx = blockIdx.x * 256 + threadIdx.x;      // < 4096*16*256
  const int j = idx & 255;
  const int mh = idx >> 8;
  const _Float16 v = (_Float16)kvu[(size_t)mh * 256 + j];
  if (j < 128) kf[(size_t)mh * 192 + j] = v;
  else         vh[(size_t)mh * 128 + (j - 128)] = v;
}

// ---------------------------------------------------------------------------
// Flash attention: block = (b,h) x 128 q rows, 8 waves x 16 rows each.
// Q frags in registers; K tiles async-DMA'd to LDS, V stored transposed.
// ---------------------------------------------------------------------------
__global__ __launch_bounds__(256)
void attn_flash(const _Float16* __restrict__ qf, const _Float16* __restrict__ kf,
                const _Float16* __restrict__ vh, _Float16* __restrict__ ao,
                const int* __restrict__ startp) {
  const int q0 = blockIdx.x * 128;
  const int bh = blockIdx.y, b = bh >> 4, h = bh & 15;
  const int t = threadIdx.x, lane = t & 31, wid = t >> 5;
  const int half = lane >> 4, lm = lane & 15;
  const int start = startp[0];

  __shared__ __align__(16) _Float16 Ks[2][32 * 200];   // [key][feat0..191] (+pad)
  __shared__ __align__(16) _Float16 Vt[2][128 * 40];   // [vcol][key] transposed
  __shared__ __align__(16) _Float16 Ps[8][16 * 40];    // per-wave P tile

  // --- Q fragments (wave's 16 rows x 192 features, A-layout) ---
  v16h qfrag[6];
  {
    const _Float16* qrow =
        qf + ((size_t)((b * 1024 + q0 + wid * 16 + lm) * 16 + h)) * 192;
#pragma unroll
    for (int kk = 0; kk < 6; ++kk)
      qfrag[kk] = ld_frag_contig(qrow + kk * 32 + half * 16);
  }

  int kmax = start + q0 + 128; if (kmax > 1024) kmax = 1024;
  const int ntiles = (kmax + 31) >> 5;

  auto load_kv = [&](int kt, int buf) {
    const int kbase = kt * 32;
    for (int c = t; c < 32 * 24; c += 256) {           // K tile: 32 x 192 halfs
      const int key = c / 24, cc = c % 24;
      const _Float16* gsrc =
          kf + ((size_t)((b * 1024 + kbase + key) * 16 + h)) * 192 + cc * 8;
#if USE_ASYNC_COPY
      ASYNC_G2L_B128(gsrc, &Ks[buf][key * 200 + cc * 8]);
#else
      uint4 d = *(const uint4*)gsrc;
      *(uint4*)(&Ks[buf][key * 200 + cc * 8]) = d;
#endif
    }
    for (int e = t; e < 32 * 128; e += 256) {          // V tile transposed
      const int vcol = e & 127, key = e >> 7;
      Vt[buf][vcol * 40 + key] =
          vh[((size_t)(b * 1024 + kbase + key)) * 2048 + h * 128 + vcol];
    }
  };

  load_kv(0, 0);

  float mrow[8], lrow[8];
  v8f oacc[8];
#pragma unroll
  for (int i = 0; i < 8; ++i) { mrow[i] = -3.0e38f; lrow[i] = 0.f; oacc[i] = zero_v8f(); }

  int buf = 0;
  for (int kt = 0; kt < ntiles; ++kt) {
    WAIT_ASYNC();
    __syncthreads();
    if (kt + 1 < ntiles) load_kv(kt + 1, buf ^ 1);
    const int kbase = kt * 32;

    // scores: 16 rows x 32 keys (two 16x16 WMMA tiles, K=192 in 6 steps)
    v8f s0 = zero_v8f(), s1 = zero_v8f();
#pragma unroll
    for (int kk = 0; kk < 6; ++kk) {
      v16h b0 = ld_frag_contig(&Ks[buf][lm * 200        + kk * 32 + half * 16]);
      v16h b1 = ld_frag_contig(&Ks[buf][(16 + lm) * 200 + kk * 32 + half * 16]);
      s0 = WMMA_F32_F16(qfrag[kk], b0, s0);
      s1 = WMMA_F32_F16(qfrag[kk], b1, s1);
    }

    // causal mask + online softmax (scale already folded into qf)
    const int qrow0 = start + q0 + wid * 16;
#pragma unroll
    for (int i = 0; i < 8; ++i) {
      const int qpos = qrow0 + i + half * 8;
      if (qpos < kbase + lm)      s0[i] = -1e9f;
      if (qpos < kbase + 16 + lm) s1[i] = -1e9f;
    }
#pragma unroll
    for (int i = 0; i < 8; ++i) {
      float mt = fmaxf(s0[i], s1[i]);
#pragma unroll
      for (int d = 1; d < 16; d <<= 1) mt = fmaxf(mt, __shfl_xor(mt, d));
      const float mnew = fmaxf(mrow[i], mt);
      const float corr = __expf(mrow[i] - mnew);
      const float p0 = __expf(s0[i] - mnew);
      const float p1 = __expf(s1[i] - mnew);
      float ps = p0 + p1;
#pragma unroll
      for (int d = 1; d < 16; d <<= 1) ps += __shfl_xor(ps, d);
      lrow[i] = lrow[i] * corr + ps;
      mrow[i] = mnew;
#pragma unroll
      for (int of = 0; of < 8; ++of) oacc[of][i] *= corr;
      Ps[wid][(i + half * 8) * 40 + lm]      = (_Float16)p0;
      Ps[wid][(i + half * 8) * 40 + 16 + lm] = (_Float16)p1;
    }
    // P is per-wave in LDS: drain our own ds stores before fragment reads
    __asm__ volatile("s_wait_dscnt 0x0" ::: "memory");

    // PV accumulation: out(16x128) += P(16x32) x V(32x128)
    v16h pa = ld_frag_contig(&Ps[wid][lm * 40 + half * 16]);
#pragma unroll
    for (int of = 0; of < 8; ++of) {
      v16h vb = ld_frag_contig(&Vt[buf][(of * 16 + lm) * 40 + half * 16]);
      oacc[of] = WMMA_F32_F16(pa, vb, oacc[of]);
    }
    buf ^= 1;
  }

  // normalize + store f16 [m][h*128 + col]
#pragma unroll
  for (int i = 0; i < 8; ++i) {
    const float inv = 1.0f / lrow[i];
    const int mg = b * 1024 + q0 + wid * 16 + i + half * 8;
    const size_t base = (size_t)mg * 2048 + h * 128 + lm;
#pragma unroll
    for (int of = 0; of < 8; ++of)
      ao[base + of * 16] = (_Float16)(oacc[of][i] * inv);
  }
}

// ---------------------------------------------------------------------------
// Host launch
// ---------------------------------------------------------------------------
extern "C" void kernel_launch(void* const* d_in, const int* in_sizes, int n_in,
                              void* d_out, int out_size, void* d_ws, size_t ws_size,
                              hipStream_t stream) {
  (void)in_sizes; (void)n_in; (void)out_size; (void)ws_size;
  constexpr int M = 4096;        // B*S
  constexpr int D = 2048, QL = 1536, H = 16, QKH = 192, KVL = 512, ROPE = 64, VH = 128;
  constexpr int NQ = H * QKH;          // 3072
  constexpr int NKV = KVL + ROPE;      // 576
  constexpr int NKVU = H * (128 + VH); // 4096

  const float* X    = (const float*)d_in[0];
  const float* bfrq = (const float*)d_in[1];
  const float* Wqd  = (const float*)d_in[2];
  const float* bqd  = (const float*)d_in[3];
  const float* gq   = (const float*)d_in[4];
  const float* Wqu  = (const float*)d_in[5];
  const float* bqu  = (const float*)d_in[6];
  const float* Wkv  = (const float*)d_in[7];
  const float* bkv  = (const float*)d_in[8];
  const float* gkv  = (const float*)d_in[9];
  const float* Wkvu = (const float*)d_in[10];
  const float* bkvu = (const float*)d_in[11];
  const float* Wo   = (const float*)d_in[12];
  const float* bo   = (const float*)d_in[13];
  const int*   start= (const int*)d_in[14];
  float* out = (float*)d_out;

  char* w = (char*)d_ws;
  auto take = [&](size_t bytes) {
    char* p = w;
    w += (bytes + 255) & ~(size_t)255;
    return p;
  };
  _Float16* Xh     = (_Float16*)take((size_t)M * D * 2);
  _Float16* Wqd_h  = (_Float16*)take((size_t)QL * D * 2);
  _Float16* Wqu_h  = (_Float16*)take((size_t)NQ * QL * 2);
  _Float16* Wkv_h  = (_Float16*)take((size_t)NKV * D * 2);
  _Float16* Wkvu_h = (_Float16*)take((size_t)NKVU * KVL * 2);
  _Float16* Wo_h   = (_Float16*)take((size_t)D * (H * VH) * 2);
  float*    qd     = (float*)take((size_t)M * QL * 4);
  _Float16* qn_h   = (_Float16*)take((size_t)M * QL * 2);
  float*    q32    = (float*)take((size_t)M * NQ * 4);
  float*    kv32   = (float*)take((size_t)M * NKV * 4);
  _Float16* kvn_h  = (_Float16*)take((size_t)M * KVL * 2);
  float*    kvu32  = (float*)take((size_t)M * NKVU * 4);
  _Float16* qf_h   = (_Float16*)take((size_t)M * H * QKH * 2);
  _Float16* kf_h   = (_Float16*)take((size_t)M * H * QKH * 2);
  _Float16* v_h    = (_Float16*)take((size_t)M * H * VH * 2);
  _Float16* ao_h   = (_Float16*)take((size_t)M * H * VH * 2);

  auto cvt = [&](const float* s, _Float16* d, int n) {
    cvt_f32_f16<<<(n + 255) / 256, 256, 0, stream>>>(s, d, n);
  };
  cvt(X,    Xh,     M * D);
  cvt(Wqd,  Wqd_h,  QL * D);
  cvt(Wqu,  Wqu_h,  NQ * QL);
  cvt(Wkv,  Wkv_h,  NKV * D);
  cvt(Wkvu, Wkvu_h, NKVU * KVL);
  cvt(Wo,   Wo_h,   D * (H * VH));

  auto gemm = [&](const _Float16* A, const _Float16* B, const float* bias,
                  float* C, int N, int K) {
    dim3 grid((N + 127) / 128, M / 128);
    gemm_wmma<<<grid, 256, 0, stream>>>(A, B, bias, C, M, N, K);
  };

  gemm(Xh, Wqd_h, bqd, qd, QL, D);                               // q down
  rms_rows<<<M, 256, 0, stream>>>(qd, gq, qn_h, QL);             // rms_q
  gemm(qn_h, Wqu_h, bqu, q32, NQ, QL);                           // q up
  rope_q<<<(M * H * QKH) / 256, 256, 0, stream>>>(q32, bfrq, qf_h);
  gemm(Xh, Wkv_h, bkv, kv32, NKV, D);                            // kv down
  kv_post<<<M, 256, 0, stream>>>(kv32, gkv, bfrq, kvn_h, kf_h);  // rms + kRot
  gemm(kvn_h, Wkvu_h, bkvu, kvu32, NKVU, KVL);                   // kv up
  build_kv<<<(M * H * 256) / 256, 256, 0, stream>>>(kvu32, kf_h, v_h);
  attn_flash<<<dim3(8, 64), 256, 0, stream>>>(qf_h, kf_h, v_h, ao_h, start);
  gemm(ao_h, Wo_h, bo, out, D, H * VH);                          // output proj
}
